// GAT_40252433498736
// MI455X (gfx1250) — matrix operand order repaired and checked by reading, not compiled, not run on hardware
//
#include <hip/hip_runtime.h>

// ---------------- CDNA5 WMMA types ----------------
typedef __attribute__((ext_vector_type(16))) __bf16 v16bf;
typedef __attribute__((ext_vector_type(8)))  float  v8f;

union ABu { uint4 u[2]; v16bf v; };

// ---------------- helpers ----------------
__device__ __forceinline__ unsigned short f2bf(float x) {
  unsigned u = __float_as_uint(x);
  unsigned r = u + 0x7FFFu + ((u >> 16) & 1u);   // round-to-nearest-even
  return (unsigned short)(r >> 16);
}
__device__ __forceinline__ unsigned fenc(float f) {   // order-preserving float->uint
  unsigned u = __float_as_uint(f);
  return (u & 0x80000000u) ? ~u : (u | 0x80000000u);
}
__device__ __forceinline__ float fdec(unsigned k) {
  return (k & 0x80000000u) ? __uint_as_float(k ^ 0x80000000u) : __uint_as_float(~k);
}
__device__ __forceinline__ float gelu_exact(float v) {
  return 0.5f * v * (1.0f + erff(v * 0.70710678118654752f));
}

// ---------------- precision prep ----------------
__global__ void k_f32_to_bf16(const float* __restrict__ in,
                              unsigned short* __restrict__ out, long n) {
  long i = (long)blockIdx.x * blockDim.x + threadIdx.x;
  if (i < n) out[i] = f2bf(in[i]);
}

// W [K, Nn] row-major f32  ->  Wop[kb][n][t] bf16, t = k & 31 (B-operand layout)
__global__ void k_wprep(const float* __restrict__ W,
                        unsigned short* __restrict__ Wop, int K, int Nn) {
  int i = blockIdx.x * blockDim.x + threadIdx.x;
  if (i >= K * Nn) return;
  int k = i / Nn, n = i % Nn;
  int kb = k >> 5, t = k & 31;
  Wop[((size_t)kb * Nn + n) * 32 + t] = f2bf(W[(size_t)k * Nn + n]);
}

// ---------------- WMMA GEMM: C[M,Nn] = A[M,K](bf16) * Wop(bf16), f32 accum ----------------
// One wave -> 16x64 output tile (4 accumulators, A fragment reused across 4 N-subtiles).
// Requires M % 16 == 0, K % 32 == 0, Nn % 64 == 0.
__global__ void __launch_bounds__(32) k_gemm_wmma(
    const unsigned short* __restrict__ A, const unsigned short* __restrict__ Bop,
    float* __restrict__ C, int K, int Nn) {
  const int lane = threadIdx.x & 31;
  const int row  = lane & 15;        // M row for A / N col for B
  const int half = lane >> 4;
  const int m0 = blockIdx.x * 16;
  const int n0 = blockIdx.y * 64;
  v8f acc0 = {}, acc1 = {}, acc2 = {}, acc3 = {};
  const unsigned short* Ar = A + (size_t)(m0 + row) * K;
  const int kblocks = K >> 5;
  for (int kb = 0; kb < kblocks; ++kb) {
    ABu a, b0, b1, b2, b3;
    const unsigned short* ap = Ar + kb * 32 + half * 16;
    a.u[0] = *(const uint4*)(ap);        // K = kb*32 + half*16 + [0..7]
    a.u[1] = *(const uint4*)(ap + 8);    // K = kb*32 + half*16 + 8 + [0..7]
    const unsigned short* bp =
        Bop + ((size_t)kb * Nn + (n0 + row)) * 32 + half * 16;
    b0.u[0] = *(const uint4*)(bp);                 // n-subtile 0
    b0.u[1] = *(const uint4*)(bp + 8);
    b1.u[0] = *(const uint4*)(bp + 16 * 32);       // n-subtile 1
    b1.u[1] = *(const uint4*)(bp + 16 * 32 + 8);
    b2.u[0] = *(const uint4*)(bp + 32 * 32);       // n-subtile 2
    b2.u[1] = *(const uint4*)(bp + 32 * 32 + 8);
    b3.u[0] = *(const uint4*)(bp + 48 * 32);       // n-subtile 3
    b3.u[1] = *(const uint4*)(bp + 48 * 32 + 8);
    acc0 = __builtin_amdgcn_wmma_f32_16x16x32_bf16(false, a.v, false, b0.v,
                                                   (short)0, acc0, false, false);
    acc1 = __builtin_amdgcn_wmma_f32_16x16x32_bf16(false, a.v, false, b1.v,
                                                   (short)0, acc1, false, false);
    acc2 = __builtin_amdgcn_wmma_f32_16x16x32_bf16(false, a.v, false, b2.v,
                                                   (short)0, acc2, false, false);
    acc3 = __builtin_amdgcn_wmma_f32_16x16x32_bf16(false, a.v, false, b3.v,
                                                   (short)0, acc3, false, false);
  }
  float* Cr = C + (size_t)m0 * Nn + n0 + row;
#pragma unroll
  for (int r = 0; r < 8; ++r) {
    float* cro = Cr + (size_t)(r + 8 * half) * Nn;
    cro[0]  = acc0[r];
    cro[16] = acc1[r];
    cro[32] = acc2[r];
    cro[48] = acc3[r];
  }
}

// ---------------- attention scores: a_src/a_dst [N,H] ----------------
__global__ void k_att(const float* __restrict__ xp, const float* __restrict__ att_s,
                      const float* __restrict__ att_d, float* __restrict__ as_,
                      float* __restrict__ ad_, int N, int H, int Cc) {
  int i = blockIdx.x * blockDim.x + threadIdx.x;
  if (i >= N * H) return;
  int n = i / H, h = i % H;
  const float* xr = xp + ((size_t)n * H + h) * Cc;
  float s = 0.f, d = 0.f;
  for (int c = 0; c < Cc; ++c) {
    float v = xr[c];
    s += v * att_s[h * Cc + c];
    d += v * att_d[h * Cc + c];
  }
  as_[i] = s; ad_[i] = d;
}

__device__ __forceinline__ float edge_val(const int* __restrict__ src,
                                          const int* __restrict__ dst,
                                          const float* __restrict__ as_,
                                          const float* __restrict__ ad_,
                                          long e, long E, int H, int h,
                                          int& s_out, int& d_out) {
  int s, d;
  if (e < E) { s = src[e]; d = dst[e]; } else { s = d = (int)(e - E); }  // self loops
  float v = as_[(size_t)s * H + h] + ad_[(size_t)d * H + h];
  v = v > 0.f ? v : 0.2f * v;                                            // LeakyReLU(0.2)
  s_out = s; d_out = d;
  return v;
}

// ---------------- segment max over dst ----------------
__global__ void k_edge_max(const int* __restrict__ src, const int* __restrict__ dst,
                           const float* __restrict__ as_, const float* __restrict__ ad_,
                           unsigned* __restrict__ emax, long E, long Et, int H) {
  long i = (long)blockIdx.x * blockDim.x + threadIdx.x;
  if (i >= Et * H) return;
  long e = i / H; int h = (int)(i % H);
  int s, d;
  float v = edge_val(src, dst, as_, ad_, e, E, H, h, s, d);
  atomicMax(&emax[(size_t)d * H + h], fenc(v));
}

// ---------------- p = exp(e - emax[dst]); denom = segment_sum(p) ----------------
__global__ void k_edge_p(const int* __restrict__ src, const int* __restrict__ dst,
                         const float* __restrict__ as_, const float* __restrict__ ad_,
                         const unsigned* __restrict__ emax, float* __restrict__ p,
                         float* __restrict__ denom, long E, long Et, int H) {
  long i = (long)blockIdx.x * blockDim.x + threadIdx.x;
  if (i >= Et * H) return;
  long e = i / H; int h = (int)(i % H);
  int s, d;
  float v = edge_val(src, dst, as_, ad_, e, E, H, h, s, d);
  float pv = expf(v - fdec(emax[(size_t)d * H + h]));
  p[i] = pv;
  atomicAdd(&denom[(size_t)d * H + h], pv);
}

// ---------------- out[dst] += xp[src] * alpha ----------------
__global__ void k_edge_aggr(const int* __restrict__ src, const int* __restrict__ dst,
                            const float* __restrict__ xp, const float* __restrict__ p,
                            const float* __restrict__ denom, float* __restrict__ out,
                            long E, long Et, int H, int Cc) {
  const long HC = (long)H * Cc;
  long i = (long)blockIdx.x * blockDim.x + threadIdx.x;
  if (i >= Et * HC) return;
  long e = i / HC; int j = (int)(i % HC); int h = j / Cc;
  int s, d;
  if (e < E) { s = src[e]; d = dst[e]; } else { s = d = (int)(e - E); }
  float alpha = p[e * H + h] / (denom[(size_t)d * H + h] + 1e-16f);
  atomicAdd(&out[(size_t)d * HC + j], xp[(size_t)s * HC + j] * alpha);
}

// ---------------- bias + GELU, write bf16 activations for next GEMM ----------------
__global__ void k_post_concat(const float* __restrict__ aggr, const float* __restrict__ bias,
                              unsigned short* __restrict__ hbf, long N, int HC) {
  long i = (long)blockIdx.x * blockDim.x + threadIdx.x;
  if (i >= N * (long)HC) return;
  int j = (int)(i % HC);
  hbf[i] = f2bf(gelu_exact(aggr[i] + bias[j]));
}

// ---------------- head mean + bias + GELU (layer 2, concat=False) ----------------
__global__ void k_post_mean(const float* __restrict__ aggr, const float* __restrict__ bias,
                            float* __restrict__ feat, long N, int H, int Cc) {
  long i = (long)blockIdx.x * blockDim.x + threadIdx.x;
  if (i >= N * (long)Cc) return;
  long n = i / Cc; int c = (int)(i % Cc);
  float v = 0.f;
  for (int h = 0; h < H; ++h) v += aggr[((size_t)n * H + h) * Cc + c];
  feat[i] = gelu_exact(v * (1.0f / H) + bias[c]);
}

// ---------------- global mean pool over batch ----------------
__global__ void k_pool_add(const float* __restrict__ feat, const int* __restrict__ batch,
                           float* __restrict__ psum, float* __restrict__ pcnt,
                           long N, int Cc) {
  long i = (long)blockIdx.x * blockDim.x + threadIdx.x;
  if (i >= N * (long)Cc) return;
  long n = i / Cc; int c = (int)(i % Cc);
  int b = batch[n];
  atomicAdd(&psum[(size_t)b * Cc + c], feat[i]);
  if (c == 0) atomicAdd(&pcnt[b], 1.0f);
}

__global__ void k_pool_div(const float* __restrict__ psum, const float* __restrict__ pcnt,
                           float* __restrict__ out, int B, int Cc) {
  int i = blockIdx.x * blockDim.x + threadIdx.x;
  if (i >= B * Cc) return;
  out[i] = psum[i] / fmaxf(pcnt[i / Cc], 1.0f);
}

// ---------------- host ----------------
extern "C" void kernel_launch(void* const* d_in, const int* in_sizes, int n_in,
                              void* d_out, int out_size, void* d_ws, size_t ws_size,
                              hipStream_t stream) {
  (void)n_in; (void)out_size; (void)ws_size;
  const float* x     = (const float*)d_in[0];
  const int*   ei    = (const int*)d_in[1];
  const int*   batch = (const int*)d_in[2];
  const float* Ws[3] = {(const float*)d_in[3],  (const float*)d_in[7],  (const float*)d_in[11]};
  const float* As[3] = {(const float*)d_in[4],  (const float*)d_in[8],  (const float*)d_in[12]};
  const float* Ad[3] = {(const float*)d_in[5],  (const float*)d_in[9],  (const float*)d_in[13]};
  const float* Bs[3] = {(const float*)d_in[6],  (const float*)d_in[10], (const float*)d_in[14]};

  const long N  = in_sizes[0] / 128;   // 50000 (divisible by 16)
  const long E  = in_sizes[1] / 2;     // 400000
  const long Et = E + N;               // + self loops
  const int  H  = 4;
  const int  Karr[3] = {128, 64, 256};
  const int  Carr[3] = {16, 64, 64};
  const int* src = ei;
  const int* dst = ei + E;

  // carve workspace (256B aligned chunks)
  size_t off = 0;
  auto carve = [&](size_t bytes) -> void* {
    void* p = (char*)d_ws + off;
    off += (bytes + 255) & ~(size_t)255;
    return p;
  };
  unsigned short* hbf  = (unsigned short*)carve((size_t)N * 256 * 2); // bf16 activations
  unsigned short* wop  = (unsigned short*)carve((size_t)256 * 256 * 2);
  float*    xp    = (float*)carve((size_t)N * 256 * 4);
  float*    aggr  = (float*)carve((size_t)N * 256 * 4);
  float*    as_   = (float*)carve((size_t)N * H * 4);
  float*    ad_   = (float*)carve((size_t)N * H * 4);
  unsigned* emax  = (unsigned*)carve((size_t)N * H * 4);
  float*    denom = (float*)carve((size_t)N * H * 4);
  float*    pbuf  = (float*)carve((size_t)Et * H * 4);
  float*    psum  = (float*)carve((size_t)64 * 64 * 4);
  float*    pcnt  = (float*)carve((size_t)64 * 4);

  const int TB = 256;
  auto cdiv = [](long a, long b) -> unsigned { return (unsigned)((a + b - 1) / b); };

  // input -> bf16
  k_f32_to_bf16<<<cdiv(N * 128, TB), TB, 0, stream>>>(x, hbf, N * 128);

  for (int l = 0; l < 3; ++l) {
    const int K = Karr[l], Cc = Carr[l], HC = H * Cc;
    // weight swizzle to B-operand layout
    k_wprep<<<cdiv((long)K * HC, TB), TB, 0, stream>>>(Ws[l], wop, K, HC);
    // xp = h @ W  (bf16 WMMA, f32 accumulate); each wave = 16x64 tile
    dim3 g((unsigned)(N / 16), (unsigned)(HC / 64));
    k_gemm_wmma<<<g, 32, 0, stream>>>(hbf, wop, xp, K, HC);
    // attention logits per node/head
    k_att<<<cdiv(N * H, TB), TB, 0, stream>>>(xp, As[l], Ad[l], as_, ad_, (int)N, H, Cc);
    // segment softmax + aggregation
    hipMemsetAsync(emax, 0, (size_t)N * H * 4, stream);   // fenc-encoded -inf floor
    hipMemsetAsync(denom, 0, (size_t)N * H * 4, stream);
    hipMemsetAsync(aggr, 0, (size_t)N * HC * 4, stream);
    k_edge_max<<<cdiv(Et * H, TB), TB, 0, stream>>>(src, dst, as_, ad_, emax, E, Et, H);
    k_edge_p<<<cdiv(Et * H, TB), TB, 0, stream>>>(src, dst, as_, ad_, emax, pbuf, denom, E, Et, H);
    k_edge_aggr<<<cdiv(Et * (long)HC, TB), TB, 0, stream>>>(src, dst, xp, pbuf, denom, aggr,
                                                            E, Et, H, Cc);
    if (l < 2) {
      k_post_concat<<<cdiv(N * (long)HC, TB), TB, 0, stream>>>(aggr, Bs[l], hbf, N, HC);
    } else {
      k_post_mean<<<cdiv(N * (long)Cc, TB), TB, 0, stream>>>(aggr, Bs[l], xp, N, H, Cc);
    }
  }

  // global mean pool over 64 graphs
  hipMemsetAsync(psum, 0, (size_t)64 * 64 * 4, stream);
  hipMemsetAsync(pcnt, 0, (size_t)64 * 4, stream);
  k_pool_add<<<cdiv(N * 64, TB), TB, 0, stream>>>(xp, batch, psum, pcnt, N, 64);
  k_pool_div<<<cdiv(64 * 64, TB), TB, 0, stream>>>(psum, pcnt, (float*)d_out, 64, 64);
}
